// DGM_c_58308476011168
// MI455X (gfx1250) — matrix operand order repaired and checked by reading, not compiled, not run on hardware
//
#include <hip/hip_runtime.h>
#include <hip/hip_bf16.h>

#define B_ 2
#define N_ 2048
#define FIN_ 128
#define FE_ 64
#define BN_ (B_ * N_)           // 4096
#define NE_ (N_ * FE_)          // 131072
#define BNN_ ((size_t)B_ * N_ * N_)

typedef __attribute__((ext_vector_type(16))) _Float16 v16h;
typedef __attribute__((ext_vector_type(8)))  _Float16 v8h;
typedef __attribute__((ext_vector_type(4)))  _Float16 v4h;
typedef __attribute__((ext_vector_type(8)))  float    v8f;
typedef __attribute__((ext_vector_type(4)))  float    v4f;

static __device__ __forceinline__ v16h cat16(v8h lo, v8h hi) {
    v16h r;
#pragma unroll
    for (int i = 0; i < 8; ++i) { r[i] = lo[i]; r[i + 8] = hi[i]; }
    return r;
}

static __device__ __forceinline__ v8f wmma_f16(v16h a, v16h b, v8f c) {
    return __builtin_amdgcn_wmma_f32_16x16x32_f16(
        /*neg_a=*/false, a, /*neg_b=*/false, b,
        /*c_mod=*/(short)0, c, /*reuse_a=*/false, /*reuse_b=*/false);
}

// ---------------------------------------------------------------------------
// prep: WT[e][k] = (f16) W[k][e]   (64 x 128), zero maxabs cell
// ---------------------------------------------------------------------------
__global__ void k_prep(const float* __restrict__ W, _Float16* __restrict__ WT,
                       unsigned* __restrict__ maxbits) {
    int g = blockIdx.x * 256 + threadIdx.x;        // [0, 8192)
    int e = g & 63, k = g >> 6;                    // coalesced read of W
    WT[e * FIN_ + k] = (_Float16)W[g];
    if (g == 0) *maxbits = 0u;
}

// ---------------------------------------------------------------------------
// t = x @ W  -> stored transposed f16: tT[b][e][n]
// grid: BN/32 blocks, 256 thr (8 waves): wave = (rowtile rt<2) x (coltile ct<4)
// ---------------------------------------------------------------------------
__global__ void k_xw(const float* __restrict__ x, const _Float16* __restrict__ WT,
                     _Float16* __restrict__ tT) {
    int w = threadIdx.x >> 5, lane = threadIdx.x & 31;
    int hg = lane >> 4, l15 = lane & 15;
    int rt = w >> 2, ct = w & 3;
    int row0 = blockIdx.x * 32 + rt * 16;          // flat row in [0, B*N)
    int e = ct * 16 + l15;
    const float*    xrow = x + (size_t)(row0 + l15) * FIN_;
    const _Float16* wrow = WT + (size_t)e * FIN_;
    v8f c = {};
#pragma unroll
    for (int k0 = 0; k0 < FIN_; k0 += 32) {
        v4f f0 = *(const v4f*)(xrow + k0 + hg * 8);
        v4f f1 = *(const v4f*)(xrow + k0 + hg * 8 + 4);
        v4f f2 = *(const v4f*)(xrow + k0 + 16 + hg * 8);
        v4f f3 = *(const v4f*)(xrow + k0 + 16 + hg * 8 + 4);
        v16h a;
#pragma unroll
        for (int i = 0; i < 4; ++i) {
            a[i] = (_Float16)f0[i]; a[4 + i]  = (_Float16)f1[i];
            a[8 + i] = (_Float16)f2[i]; a[12 + i] = (_Float16)f3[i];
        }
        v8h b0 = *(const v8h*)(wrow + k0 + hg * 8);
        v8h b1 = *(const v8h*)(wrow + k0 + 16 + hg * 8);
        c = wmma_f16(a, cat16(b0, b1), c);
    }
#pragma unroll
    for (int r = 0; r < 8; ++r) {
        int rf = row0 + r + 8 * hg;                // flat row
        int b  = rf >> 11, n = rf & (N_ - 1);
        tT[((size_t)b * FE_ + e) * N_ + n] = (_Float16)c[r];
    }
}

// ---------------------------------------------------------------------------
// x_emb = relu(A @ t + bias)   A: f32 [b][N][N], tT: f16 [b][FE][N]
// grid: (N/32, B), 256 thr. LDS-staged A tile (f32->f16, padded stride 40).
// ---------------------------------------------------------------------------
__global__ void k_embed(const float* __restrict__ A, const _Float16* __restrict__ tT,
                        const float* __restrict__ bias, float* __restrict__ x_emb) {
    __shared__ _Float16 As[32 * 40];
    int b = blockIdx.y;
    int n0 = blockIdx.x * 32;
    int tid = threadIdx.x;
    int w = tid >> 5, lane = tid & 31, hg = lane >> 4, l15 = lane & 15;
    int rt = w >> 2, ct = w & 3;
    const float*    Ab = A + ((size_t)b * N_ + n0) * N_;
    const _Float16* br = tT + ((size_t)b * FE_ + ct * 16 + l15) * N_;
    const _Float16* ar = As + (rt * 16 + l15) * 40;
    int lrow = tid >> 3, lcol = (tid & 7) * 4;     // A-tile staging assignment
    v8f c = {};
    for (int k0 = 0; k0 < N_; k0 += 32) {
        __syncthreads();
        v4f av = *(const v4f*)(Ab + (size_t)lrow * N_ + k0 + lcol);
        v4h ah; ah[0] = (_Float16)av[0]; ah[1] = (_Float16)av[1];
        ah[2] = (_Float16)av[2]; ah[3] = (_Float16)av[3];
        *(v4h*)(As + lrow * 40 + lcol) = ah;
        __syncthreads();
        v8h a0 = *(const v8h*)(ar + hg * 8);
        v8h a1 = *(const v8h*)(ar + 16 + hg * 8);
        v8h b0 = *(const v8h*)(br + k0 + hg * 8);
        v8h b1 = *(const v8h*)(br + k0 + 16 + hg * 8);
        c = wmma_f16(cat16(a0, a1), cat16(b0, b1), c);
    }
    float bv = bias[ct * 16 + l15];
#pragma unroll
    for (int r = 0; r < 8; ++r) {
        float v = c[r] + bv;
        v = v > 0.f ? v : 0.f;
        x_emb[(((size_t)b * N_) + n0 + rt * 16 + r + 8 * hg) * FE_ + ct * 16 + l15] = v;
    }
}

// ---------------------------------------------------------------------------
// centroid[b][e] = mean_n x_emb[b][n][e]    grid: B*64 blocks
// ---------------------------------------------------------------------------
__global__ void k_centroid(const float* __restrict__ x_emb, float* __restrict__ centroid) {
    int be = blockIdx.x, b = be >> 6, e = be & 63;
    const float* p = x_emb + (size_t)b * NE_ + e;
    float s = 0.f;
    for (int n = threadIdx.x; n < N_; n += 256) s += p[(size_t)n * FE_];
    __shared__ float red[256];
    red[threadIdx.x] = s;
    __syncthreads();
    for (int off = 128; off; off >>= 1) {
        if (threadIdx.x < off) red[threadIdx.x] += red[threadIdx.x + off];
        __syncthreads();
    }
    if (threadIdx.x == 0) centroid[be] = red[0] * (1.f / (float)N_);
}

// ---------------------------------------------------------------------------
// global max |x_emb - centroid| via block reduce + atomicMax on float bits
// grid: 512 blocks x 256 thr x 2 elems = 262144
// ---------------------------------------------------------------------------
__global__ void k_maxabs(const float* __restrict__ x_emb, const float* __restrict__ centroid,
                         unsigned* __restrict__ maxbits) {
    size_t i0 = ((size_t)blockIdx.x * 256 + threadIdx.x) * 2;
    float m = 0.f;
#pragma unroll
    for (int j = 0; j < 2; ++j) {
        size_t i = i0 + j;
        int e = (int)(i & 63);
        int b = (int)(i >> 17);                    // N*FE = 2^17
        float v = fabsf(x_emb[i] - centroid[b * 64 + e]);
        m = fmaxf(m, v);
    }
    __shared__ float red[256];
    red[threadIdx.x] = m;
    __syncthreads();
    for (int off = 128; off; off >>= 1) {
        if (threadIdx.x < off)
            red[threadIdx.x] = fmaxf(red[threadIdx.x], red[threadIdx.x + off]);
        __syncthreads();
    }
    if (threadIdx.x == 0) atomicMax(maxbits, __float_as_uint(red[0]));
}

// ---------------------------------------------------------------------------
// xs = (x_emb - centroid)*scale  (f16, row-major [BN][64]); sq[row] = ||xs||^2
// one wave32 per row; grid 512 x 256
// ---------------------------------------------------------------------------
__global__ void k_xs(const float* __restrict__ x_emb, const float* __restrict__ centroid,
                     const unsigned* __restrict__ maxbits,
                     _Float16* __restrict__ xs, float* __restrict__ sq) {
    int w = threadIdx.x >> 5, lane = threadIdx.x & 31;
    int row = blockIdx.x * 8 + w;                  // [0, BN)
    int b = row >> 11;
    float scale = 0.9f / __uint_as_float(*maxbits);
    const float* p = x_emb + (size_t)row * FE_;
    _Float16*    q = xs + (size_t)row * FE_;
    float acc = 0.f;
#pragma unroll
    for (int j = 0; j < 2; ++j) {
        int e = lane + 32 * j;
        float v = (p[e] - centroid[b * 64 + e]) * scale;
        q[e] = (_Float16)v;
        acc += v * v;
    }
    for (int off = 16; off; off >>= 1) acc += __shfl_xor(acc, off, 32);
    if (lane == 0) sq[row] = acc;
}

// ---------------------------------------------------------------------------
// gram = xs @ xs^T ; D = max(sq_n + sq_m - 2*gram, 0); w = sigmoid(T*(|th|-D))
// grid: (N/64 cols, N/32 rows, B), 256 thr, wave tile 16x16, K=64 (2 WMMAs)
// ---------------------------------------------------------------------------
__global__ void k_adj(const _Float16* __restrict__ xs, const float* __restrict__ sq,
                      const float* __restrict__ temp, const float* __restrict__ thr,
                      float* __restrict__ ew) {
    int b = blockIdx.z;
    int w = threadIdx.x >> 5, lane = threadIdx.x & 31, hg = lane >> 4, l15 = lane & 15;
    int rt = w >> 2, ct = w & 3;
    int n0 = blockIdx.y * 32 + rt * 16;
    int m0 = blockIdx.x * 64 + ct * 16;
    const _Float16* xb = xs + (size_t)b * NE_;
    const _Float16* ar = xb + (size_t)(n0 + l15) * FE_;
    const _Float16* brp = xb + (size_t)(m0 + l15) * FE_;   // xs row == B-operand column
    v8f c = {};
#pragma unroll
    for (int k0 = 0; k0 < FE_; k0 += 32) {
        v8h a0 = *(const v8h*)(ar + k0 + hg * 8);
        v8h a1 = *(const v8h*)(ar + k0 + 16 + hg * 8);
        v8h b0 = *(const v8h*)(brp + k0 + hg * 8);
        v8h b1 = *(const v8h*)(brp + k0 + 16 + hg * 8);
        c = wmma_f16(cat16(a0, a1), cat16(b0, b1), c);
    }
    float T = *temp, th = fabsf(*thr);
    const float* sqb = sq + b * N_;
    float sqm = sqb[m0 + l15];
    float* out = ew + (size_t)b * N_ * N_;
#pragma unroll
    for (int r = 0; r < 8; ++r) {
        int n = n0 + r + 8 * hg;
        float D = sqb[n] + sqm - 2.f * c[r];
        D = D > 0.f ? D : 0.f;
        float wgt = 1.f / (1.f + __expf(-T * (th - D)));
        out[(size_t)n * N_ + m0 + l15] = wgt;
    }
}

// ---------------------------------------------------------------------------
// edge_index: rows = idx//N ; cols = idx%N + N*(idx//(N*N)). 67 MB pure stream.
// ---------------------------------------------------------------------------
__global__ void k_eidx(float* __restrict__ rows, float* __restrict__ cols) {
    size_t i = ((size_t)blockIdx.x * 256 + threadIdx.x) * 4;
    v4f rv, cv;
#pragma unroll
    for (int j = 0; j < 4; ++j) {
        size_t idx = i + j;
        unsigned r = (unsigned)(idx >> 11);              // idx / N
        unsigned cc = (unsigned)(idx & (N_ - 1)) + ((r >> 11) << 11);  // + N*batch
        rv[j] = (float)r;
        cv[j] = (float)cc;
    }
    __builtin_nontemporal_store(rv, (v4f*)(rows + i));
    __builtin_nontemporal_store(cv, (v4f*)(cols + i));
}

// ---------------------------------------------------------------------------
extern "C" void kernel_launch(void* const* d_in, const int* in_sizes, int n_in,
                              void* d_out, int out_size, void* d_ws, size_t ws_size,
                              hipStream_t stream) {
    const float* x    = (const float*)d_in[0];
    const float* A    = (const float*)d_in[1];
    const float* W    = (const float*)d_in[2];
    const float* bias = (const float*)d_in[3];
    const float* temp = (const float*)d_in[4];
    const float* thr  = (const float*)d_in[5];

    float* out = (float*)d_out;
    float* x_emb     = out;                               // B*N*FE      = 262144
    float* eidx_rows = out + (size_t)B_ * N_ * FE_;       // B*N*N       = 8388608
    float* eidx_cols = eidx_rows + BNN_;                  // B*N*N
    float* ew        = eidx_cols + BNN_;                  // B*N*N

    // workspace layout
    _Float16* WT = (_Float16*)d_ws;                       // 64*128
    _Float16* tT = WT + FE_ * FIN_;                       // B*FE*N
    _Float16* xs = tT + (size_t)B_ * FE_ * N_;            // B*N*FE
    float*    sq = (float*)(xs + (size_t)B_ * N_ * FE_);  // B*N
    float*    centroid = sq + BN_;                        // B*FE
    unsigned* maxbits  = (unsigned*)(centroid + B_ * FE_);

    k_prep    <<<32, 256, 0, stream>>>(W, WT, maxbits);
    k_xw      <<<BN_ / 32, 256, 0, stream>>>(x, WT, tT);
    k_embed   <<<dim3(N_ / 32, B_), 256, 0, stream>>>(A, tT, bias, x_emb);
    k_centroid<<<B_ * FE_, 256, 0, stream>>>(x_emb, centroid);
    k_maxabs  <<<512, 256, 0, stream>>>(x_emb, centroid, maxbits);
    k_xs      <<<512, 256, 0, stream>>>(x_emb, centroid, maxbits, xs, sq);
    k_adj     <<<dim3(N_ / 64, N_ / 32, B_), 256, 0, stream>>>(xs, sq, temp, thr, ew);
    k_eidx    <<<(unsigned)(BNN_ / (256 * 4)), 256, 0, stream>>>(eidx_rows, eidx_cols);
}
